// SelfAttentionDecoder_84920093377175
// MI455X (gfx1250) — matrix operand order repaired and checked by reading, not compiled, not run on hardware
//
#include <hip/hip_runtime.h>
#include <hip/hip_bf16.h>
#include <stdint.h>

// ---------------------------------------------------------------------------
// Masked MHA for MI455X (gfx1250, wave32, WMMA + TDM).
//   B=16, N1=N2=1024, D=1024, H=16, DH=64
// fp32 -> bf16 once; all GEMMs via v_wmma_f32_16x16x32_bf16 (fp32 accum).
// Projection GEMMs stage W tiles into LDS with the Tensor Data Mover
// (tensor_load_to_lds, TENSORcnt double buffering). Attention keeps scores in
// per-wave LDS (320KB WGP pool) with a two-pass wave32 softmax.
// ---------------------------------------------------------------------------

typedef __attribute__((ext_vector_type(16))) __bf16 v16bf;
typedef __attribute__((ext_vector_type(8)))  float  v8f;
typedef unsigned int v4u __attribute__((ext_vector_type(4)));
typedef int          v8i __attribute__((ext_vector_type(8)));
typedef int          v4i __attribute__((ext_vector_type(4)));

union Frag16 { uint4 u4[2]; v16bf v; };   // 16 bf16 = one WMMA A/B operand
union BF16x8 { uint4 u4; __bf16 b[8]; };
union BF16x4 { uint2 u2; __bf16 b[4]; };

static __device__ __forceinline__ v8f wmma_bf16(v16bf a, v16bf b, v8f c) {
  // 8-arg form: (neg_a, A, neg_b, B, c_mod, C, reuse_a, reuse_b)
  return __builtin_amdgcn_wmma_f32_16x16x32_bf16(false, a, false, b,
                                                 (short)0, c, false, false);
}

// LDS byte offset of a generic pointer: integer value of its addrspace(3) view
typedef __attribute__((address_space(3))) char lds_char;
static __device__ __forceinline__ unsigned lds_offset(void* p) {
  return (unsigned)(unsigned long long)(lds_char*)p;
}

// ---------------------------------------------------------------------------
// TDM: load a 2-D bf16 tile (64 K-elems x 64 rows, row stride 1024 elems)
// from global into LDS at lds_byte_off, padding 16B after every 128B so the
// staged row stride is 144B (16 distinct LDS banks for 16-lane b128 reads).
// D# packing per CDNA5 ISA ch.8 (group0 128b, group1 256b, groups 2/3 zero).
// This toolchain exposes the 6-arg builtin (g0, g1, g2, g3, g4, cpol).
// ---------------------------------------------------------------------------
static __device__ __forceinline__ void tdm_load_w_tile(unsigned lds_byte_off,
                                                       const void* gptr) {
  unsigned long long ga = (unsigned long long)(uintptr_t)gptr;
  v4u g0;
  g0.x = 0x1u;                                   // count=1 (valid), user mode
  g0.y = lds_byte_off;                           // bits 63:32  lds_addr
  g0.z = (unsigned)(ga & 0xFFFFFFFFu);           // bits 95:64  global_addr lo
  g0.w = (unsigned)((ga >> 32) & 0x01FFFFFFu)    // bits 120:96 global_addr hi
         | 0x80000000u;                          // bits 127:126 type=2 (image)
  // group1: data_size=1 (2B) | pad_enable | pad_interval=4 (32 DW) |
  //         pad_amount=3 (4 DW); tensor dims 1<<20 (never OOB);
  //         tile 64(k) x 64(rows); tensor_dim0_stride = 1024 elems.
  v8i g1;
  g1[0] = (int)0x07110000u;
  g1[1] = (int)(((1u << 20) & 0xFFFFu) << 16);               // dim0[15:0]
  g1[2] = (int)((((1u << 20) >> 16) & 0xFFFFu)               // dim0[31:16]
                | (((1u << 20) & 0xFFFFu) << 16));           // dim1[15:0]
  g1[3] = (int)(((((1u << 20) >> 16)) & 0xFFFFu)             // dim1[31:16]
                | (64u << 16));                              // tile_dim0 = 64
  g1[4] = (int)64u;                                          // tile_dim1 = 64
  g1[5] = (int)1024u;                                        // dim0_stride lo
  g1[6] = 0;                                                 // stride hi, d1s lo
  g1[7] = 0;
  v4i gz4 = {0, 0, 0, 0};                                    // groups 2/3: 2-D
  v8i gz8 = {0, 0, 0, 0, 0, 0, 0, 0};
  __builtin_amdgcn_tensor_load_to_lds(g0, g1, gz4, gz4, gz8, 0);
}

#define LOG2E 1.4426950408889634f
#define NEGBIG (-1.0e30f)

// ---------------------------------------------------------------------------
// fp32 -> bf16 conversion (vectorized x4), grid-stride
// ---------------------------------------------------------------------------
__global__ __launch_bounds__(256) void cvt_f32_bf16(const float* __restrict__ src,
                                                    __bf16* __restrict__ dst, int n4) {
  for (int i = blockIdx.x * blockDim.x + threadIdx.x; i < n4;
       i += gridDim.x * blockDim.x) {
    float4 f = ((const float4*)src)[i];
    BF16x4 o;
    o.b[0] = (__bf16)f.x; o.b[1] = (__bf16)f.y;
    o.b[2] = (__bf16)f.z; o.b[3] = (__bf16)f.w;
    ((uint2*)dst)[i] = o.u2;
  }
}

// ---------------------------------------------------------------------------
// GEMM: out = A[16384,1024](bf16) * W[1024,1024]^T(bf16) + bias
// block = 128 threads (4 waves). wave tile = 32(M) x 64(N); block = 128 x 64.
// W tiles (64 cols x 64 K) staged in LDS by TDM, double-buffered on TENSORcnt;
// B fragments then come from ds_load_b128 and are shared by all 4 waves.
// mode 0: bf16 out, head-split    [B,H,Nseq,64]   (Q, K)
// mode 1: bf16 out, head-split^T  [B,H,64,Nseq]   (V transposed)
// mode 2: fp32 out, row-major     [16384,1024]    (final projection)
// ---------------------------------------------------------------------------
#define WROWB 144   // staged LDS row stride in bytes (128 data + 16 TDM pad)
__global__ __launch_bounds__(128)
void gemm_wmma_bf16(const __bf16* __restrict__ A, const __bf16* __restrict__ W,
                    const float* __restrict__ bias, void* __restrict__ out,
                    int mode) {
  __shared__ __align__(16) char wtile[2][64 * WROWB];   // 2 x 9216 B

  const int lane = threadIdx.x & 31;
  const int wave = threadIdx.x >> 5;
  const int half = lane >> 4;       // lane half selects K-split per ISA layout
  const int l16  = lane & 15;
  const int m_base = blockIdx.x * 128 + wave * 32;
  const int n_base = blockIdx.y * 64;

  v8f acc[2][4] = {};

  const size_t arow0 = (size_t)(m_base + l16) * 1024;
  const size_t arow1 = (size_t)(m_base + 16 + l16) * 1024;
  const size_t wbase = (size_t)n_base * 1024;

  // prologue: stage K-tile 0 into buffer 0
  if (wave == 0) {
    tdm_load_w_tile(lds_offset(&wtile[0][0]), W + wbase);
    __builtin_amdgcn_s_wait_tensorcnt(0);
  }
  __syncthreads();

  for (int kt = 0; kt < 16; ++kt) {           // 16 K-tiles of 64
    const int k0 = kt * 64;
    if (wave == 0 && kt < 15)                 // prefetch next tile via TDM
      tdm_load_w_tile(lds_offset(&wtile[(kt + 1) & 1][0]),
                      W + wbase + (size_t)(k0 + 64));

    const char* wt = &wtile[kt & 1][0];
#pragma unroll
    for (int ks = 0; ks < 64; ks += 32) {
      Frag16 a[2], b[4];
      // A fragment: lane = row (M); elems 0..7 -> +half*8, 8..15 -> +16+half*8
      a[0].u4[0] = *(const uint4*)(A + arow0 + k0 + ks + half * 8);
      a[0].u4[1] = *(const uint4*)(A + arow0 + k0 + ks + 16 + half * 8);
      a[1].u4[0] = *(const uint4*)(A + arow1 + k0 + ks + half * 8);
      a[1].u4[1] = *(const uint4*)(A + arow1 + k0 + ks + 16 + half * 8);
      __builtin_prefetch(A + arow0 + k0 + ks + 64, 0, 1);  // global_prefetch_b8
#pragma unroll
      for (int j = 0; j < 4; ++j) {
        // B fragment from staged LDS: lane = output column; K = half*8, 16+half*8
        const char* p = wt + (j * 16 + l16) * WROWB + ks * 2 + half * 16;
        b[j].u4[0] = *(const uint4*)(p);
        b[j].u4[1] = *(const uint4*)(p + 32);
      }
#pragma unroll
      for (int i = 0; i < 2; ++i)
#pragma unroll
        for (int j = 0; j < 4; ++j)
          acc[i][j] = wmma_bf16(a[i].v, b[j].v, acc[i][j]);
    }

    __syncthreads();                          // all waves done with buf[kt&1]
    if (wave == 0 && kt < 15)
      __builtin_amdgcn_s_wait_tensorcnt(0);   // next buffer landed in LDS
    __syncthreads();                          // publish to all waves
  }

  float bs[4];
#pragma unroll
  for (int j = 0; j < 4; ++j) bs[j] = bias[n_base + j * 16 + l16];

#pragma unroll
  for (int i = 0; i < 2; ++i)
#pragma unroll
    for (int j = 0; j < 4; ++j)
#pragma unroll
      for (int r = 0; r < 8; ++r) {
        int m = m_base + i * 16 + (half << 3) + r;   // C layout: M = r + half*8
        int c = n_base + j * 16 + l16;
        float v = acc[i][j][r] + bs[j];
        if (mode == 2) {
          ((float*)out)[(size_t)m * 1024 + c] = v;
        } else {
          int bb = m >> 10, n = m & 1023, h = c >> 6, dh = c & 63;
          size_t o = (mode == 0)
                         ? ((size_t)((bb * 16 + h) * 1024 + n) * 64 + dh)
                         : ((size_t)((bb * 16 + h) * 64 + dh) * 1024 + n);
          ((__bf16*)out)[o] = (__bf16)v;
        }
      }
}

// ---------------------------------------------------------------------------
// Fused masked attention: per (b,h), per 16-query tile (one wave each).
// block = 128 threads (4 waves), grid = (N1/64, H, B).
// Per-wave dynamic LDS: 16 x 1032 bf16 score/prob rows (+8 bf16 row pad) +
// 16 fp32 1/sum = 33088 B; x4 waves = 132352 B of the WGP's 320KB pool.
// ---------------------------------------------------------------------------
__global__ __launch_bounds__(128)
void attn_wmma(const __bf16* __restrict__ Qh, const __bf16* __restrict__ Kh,
               const __bf16* __restrict__ Vt, const unsigned char* __restrict__ adj,
               __bf16* __restrict__ O) {
  extern __shared__ char smem_raw[];
  const int lane = threadIdx.x & 31;
  const int wave = threadIdx.x >> 5;
  const int half = lane >> 4;
  const int l16  = lane & 15;
  const int b = blockIdx.z, h = blockIdx.y;
  const int bh = b * 16 + h;
  const int m0 = blockIdx.x * 64 + wave * 16;   // query tile base in N1

  const int ROWB = 1032;                         // bf16 elems per LDS row
  __bf16* S  = (__bf16*)(smem_raw + (size_t)wave * (16 * ROWB * 2 + 64));
  float* inv = (float*)((char*)S + 16 * ROWB * 2);

  // ---- load Q fragments for K(=DH)=64 : two 16x32 A-fragments, kept in regs
  Frag16 qa[2];
  {
    size_t qb = (size_t)(bh * 1024 + m0 + l16) * 64;
#pragma unroll
    for (int t = 0; t < 2; ++t) {
      qa[t].u4[0] = *(const uint4*)(Qh + qb + t * 32 + half * 8);
      qa[t].u4[1] = *(const uint4*)(Qh + qb + t * 32 + 16 + half * 8);
    }
  }

  // ---- phase 1: S = scale * Q K^T, masked, -> LDS (bf16)
  for (int kt = 0; kt < 64; ++kt) {
    Frag16 kb[2];
    size_t kr = (size_t)(bh * 1024 + kt * 16 + l16) * 64;
#pragma unroll
    for (int t = 0; t < 2; ++t) {
      kb[t].u4[0] = *(const uint4*)(Kh + kr + t * 32 + half * 8);
      kb[t].u4[1] = *(const uint4*)(Kh + kr + t * 32 + 16 + half * 8);
    }
    v8f sc = {};
    sc = wmma_bf16(qa[0].v, kb[0].v, sc);
    sc = wmma_bf16(qa[1].v, kb[1].v, sc);
#pragma unroll
    for (int r = 0; r < 8; ++r) {
      int mrow = (half << 3) + r;
      float s = sc[r] * 0.125f;                  // 1/sqrt(64)
      if (!adj[(size_t)(m0 + mrow) * 1024 + kt * 16 + l16]) s = NEGBIG;
      S[mrow * ROWB + kt * 16 + l16] = (__bf16)s;
    }
  }

  // ---- phase 2: softmax. lane owns (row = l16, half of 1024 cols).
  // LDS ops of one wave execute in order, so no barrier is needed.
  const int row = l16;
  const int cb  = half * 512;
  float mx = -3.0e38f;
  for (int c = 0; c < 512; c += 8) {
    BF16x8 v; v.u4 = *(const uint4*)(S + row * ROWB + cb + c);
#pragma unroll
    for (int e = 0; e < 8; ++e) mx = fmaxf(mx, (float)v.b[e]);
  }
  mx = fmaxf(mx, __shfl_xor(mx, 16));            // combine column halves
  float sum = 0.f;
  for (int c = 0; c < 512; c += 8) {
    BF16x8 v; v.u4 = *(const uint4*)(S + row * ROWB + cb + c);
    BF16x8 p;
#pragma unroll
    for (int e = 0; e < 8; ++e) {
      float pe = __builtin_amdgcn_exp2f(((float)v.b[e] - mx) * LOG2E);
      sum += pe;
      p.b[e] = (__bf16)pe;
    }
    *(uint4*)(S + row * ROWB + cb + c) = p.u4;   // unnormalized probs in place
  }
  sum += __shfl_xor(sum, 16);
  if (half == 0) inv[row] = 1.0f / sum;

  // ---- phase 3: O = P * V  (P from LDS as ready A-fragments, V^T from HBM)
  v8f acco[4] = {};
  for (int kk = 0; kk < 1024; kk += 32) {
    Frag16 pa;
    pa.u4[0] = *(const uint4*)(S + l16 * ROWB + kk + half * 8);
    pa.u4[1] = *(const uint4*)(S + l16 * ROWB + kk + 16 + half * 8);
#pragma unroll
    for (int j = 0; j < 4; ++j) {
      Frag16 vb;
      size_t vr = (size_t)(bh * 64 + j * 16 + l16) * 1024 + kk + half * 8;
      vb.u4[0] = *(const uint4*)(Vt + vr);
      vb.u4[1] = *(const uint4*)(Vt + vr + 16);
      acco[j] = wmma_bf16(pa.v, vb.v, acco[j]);
    }
  }

  // ---- epilogue: scale rows by 1/sum, write head-merged [B, N1, 1024] bf16
  float4 i0 = *(const float4*)(inv + half * 8);
  float4 i1 = *(const float4*)(inv + half * 8 + 4);
  float rv[8] = {i0.x, i0.y, i0.z, i0.w, i1.x, i1.y, i1.z, i1.w};
#pragma unroll
  for (int j = 0; j < 4; ++j)
#pragma unroll
    for (int r = 0; r < 8; ++r) {
      int mrow = (half << 3) + r;
      float o = acco[j][r] * rv[r];
      O[(size_t)(b * 1024 + m0 + mrow) * 1024 + h * 64 + j * 16 + l16] = (__bf16)o;
    }
}

// ---------------------------------------------------------------------------
// launch
// ---------------------------------------------------------------------------
extern "C" void kernel_launch(void* const* d_in, const int* in_sizes, int n_in,
                              void* d_out, int out_size, void* d_ws, size_t ws_size,
                              hipStream_t stream) {
  const float* q  = (const float*)d_in[0];
  const float* x  = (const float*)d_in[1];
  const float* Wq = (const float*)d_in[2];
  const float* bq = (const float*)d_in[3];
  const float* Wk = (const float*)d_in[4];
  const float* bk = (const float*)d_in[5];
  const float* Wv = (const float*)d_in[6];
  const float* bv = (const float*)d_in[7];
  const float* Wo = (const float*)d_in[8];
  const float* bo = (const float*)d_in[9];
  // adj_mat is jnp.bool_ (1 byte/elem raw buffer)
  const unsigned char* adj = (const unsigned char*)d_in[10];

  char* ws = (char*)d_ws;                        // needs 176,160,768 B (168 MiB)
  __bf16* qbf = (__bf16*)(ws + 0);               // 32 MiB (reused as Obf later)
  __bf16* xbf = (__bf16*)(ws + 33554432);        // 32 MiB
  __bf16* wqb = (__bf16*)(ws + 67108864);        // 2 MiB
  __bf16* wkb = (__bf16*)(ws + 69206016);
  __bf16* wvb = (__bf16*)(ws + 71303168);
  __bf16* wob = (__bf16*)(ws + 73400320);
  __bf16* Qh  = (__bf16*)(ws + 75497472);        // 32 MiB  [B,H,N1,64]
  __bf16* Kh  = (__bf16*)(ws + 109051904);       // 32 MiB  [B,H,N2,64]
  __bf16* Vt  = (__bf16*)(ws + 142606336);       // 32 MiB  [B,H,64,N2]
  __bf16* Obf = qbf;                             // alias: qbf dead after Q proj

  // bf16 conversions
  cvt_f32_bf16<<<2048, 256, 0, stream>>>(q,  qbf, 16 * 1024 * 1024 / 4);
  cvt_f32_bf16<<<2048, 256, 0, stream>>>(x,  xbf, 16 * 1024 * 1024 / 4);
  cvt_f32_bf16<<<512,  256, 0, stream>>>(Wq, wqb, 1024 * 1024 / 4);
  cvt_f32_bf16<<<512,  256, 0, stream>>>(Wk, wkb, 1024 * 1024 / 4);
  cvt_f32_bf16<<<512,  256, 0, stream>>>(Wv, wvb, 1024 * 1024 / 4);
  cvt_f32_bf16<<<512,  256, 0, stream>>>(Wo, wob, 1024 * 1024 / 4);

  // projections (M = 16384 rows, N = 1024)
  dim3 gg(128, 16, 1);
  gemm_wmma_bf16<<<gg, 128, 0, stream>>>(qbf, wqb, bq, (void*)Qh, 0);
  gemm_wmma_bf16<<<gg, 128, 0, stream>>>(xbf, wkb, bk, (void*)Kh, 0);
  gemm_wmma_bf16<<<gg, 128, 0, stream>>>(xbf, wvb, bv, (void*)Vt, 1);

  // fused masked attention; 132,352 B dynamic LDS (4 waves x 33,088 B)
  dim3 ga(16, 16, 16);
  attn_wmma<<<ga, 128, 132352, stream>>>(Qh, Kh, Vt, adj, Obf);

  // output projection -> fp32 d_out [B, N1, D]
  gemm_wmma_bf16<<<gg, 128, 0, stream>>>(Obf, wob, bo, d_out, 2);
}
